// Block_88338887344891
// MI455X (gfx1250) — compile-verified, matrix operand order checked by reading, not run on hardware
//
#include <hip/hip_runtime.h>
#include <stdint.h>

// ---------- types ----------
typedef __bf16 bf16_t;
typedef bf16_t v16bf __attribute__((ext_vector_type(16)));
typedef float  v8f   __attribute__((ext_vector_type(8)));
typedef int    v4i   __attribute__((ext_vector_type(4)));

#define B_   2
#define T_   2048
#define D_   1024
#define H_   16
#define HD_  64
#define DFF_ 4096
#define NTOK (B_ * T_)

#if defined(__gfx1250__) && __has_builtin(__builtin_amdgcn_global_load_async_to_lds_b128)
#define HAVE_ASYNC_LDS 1
#else
#define HAVE_ASYNC_LDS 0
#endif

__device__ __forceinline__ unsigned short f2bf(float f) {
  union { float f; unsigned int u; } c; c.f = f;
  unsigned int u = c.u;
  unsigned int r = u + 0x7FFFu + ((u >> 16) & 1u);  // round-to-nearest-even
  return (unsigned short)(r >> 16);
}

// 16-byte global -> LDS copy; async engine when available (ASYNCcnt tracked),
// else synchronous VGPR-staged copy.  Both require 16B-aligned src and dst.
__device__ __forceinline__ void cp16_g2l(unsigned short* ldst,
                                         const unsigned short* gsrc) {
#if HAVE_ASYNC_LDS
  __builtin_amdgcn_global_load_async_to_lds_b128(
      (__attribute__((address_space(1))) v4i*)gsrc,
      (__attribute__((address_space(3))) v4i*)ldst, 0, 0);
#else
  *reinterpret_cast<uint4*>(ldst) = *reinterpret_cast<const uint4*>(gsrc);
#endif
}

__device__ __forceinline__ void wait_async() {
#if HAVE_ASYNC_LDS
#if __has_builtin(__builtin_amdgcn_s_wait_asynccnt)
  __builtin_amdgcn_s_wait_asynccnt(0);
#else
  asm volatile("s_wait_asynccnt 0x0" ::: "memory");
#endif
#endif
}

union FragU { v16bf v; uint4 q[2]; };

// Load a 16-element bf16 WMMA fragment: p points at (row_base + ko) where
// ko = (lane/16)*8.  Elements 0..7 are K=ko..ko+7, elements 8..15 are
// K=ko+16..ko+23 (ISA 16-bit A/B fragment layout).  Both halves are 16B loads.
__device__ __forceinline__ v16bf load_frag(const unsigned short* p) {
  FragU f;
  f.q[0] = *reinterpret_cast<const uint4*>(p);
  f.q[1] = *reinterpret_cast<const uint4*>(p + 16);
  return f.v;
}

__device__ __forceinline__ v8f wmma_bf16(v16bf a, v16bf b, v8f c) {
  return __builtin_amdgcn_wmma_f32_16x16x32_bf16(false, a, false, b,
                                                 (short)0, c, false, false);
}

__device__ __forceinline__ v8f v8f_zero() {
  v8f z = {0.f, 0.f, 0.f, 0.f, 0.f, 0.f, 0.f, 0.f};
  return z;
}

// ---------- weight transpose + f32->bf16 convert : out[N][K] = in[K][N] ----------
__global__ __launch_bounds__(256)
void transpose_f32_bf16(const float* __restrict__ in,
                        unsigned short* __restrict__ out, int R, int C) {
  __shared__ float tile[32][33];
  const int c0 = blockIdx.x * 32, r0 = blockIdx.y * 32;
  const int tx = threadIdx.x, ty = threadIdx.y;   // 32 x 8
#pragma unroll
  for (int i = ty; i < 32; i += 8)
    tile[i][tx] = in[(size_t)(r0 + i) * C + c0 + tx];
  __syncthreads();
#pragma unroll
  for (int i = ty; i < 32; i += 8)
    out[(size_t)(c0 + i) * R + r0 + tx] = f2bf(tile[tx][i]);
}

// ---------- LayerNorm : f32 [rows][1024] -> bf16 ----------
__global__ __launch_bounds__(256)
void ln_kernel(const float* __restrict__ x, const float* __restrict__ w,
               const float* __restrict__ bsh, unsigned short* __restrict__ out) {
  __shared__ float red[16];
  const int row = blockIdx.x;
  const int t = threadIdx.x;                       // 256 threads, 4 floats each
  const float4 v = reinterpret_cast<const float4*>(x + (size_t)row * D_)[t];
  float s  = v.x + v.y + v.z + v.w;
  float s2 = v.x * v.x + v.y * v.y + v.z * v.z + v.w * v.w;
#pragma unroll
  for (int m = 1; m < 32; m <<= 1) {
    s  += __shfl_xor(s,  m, 32);
    s2 += __shfl_xor(s2, m, 32);
  }
  if ((t & 31) == 0) { red[t >> 5] = s; red[8 + (t >> 5)] = s2; }
  __syncthreads();
  float S = 0.f, S2 = 0.f;
#pragma unroll
  for (int i = 0; i < 8; ++i) { S += red[i]; S2 += red[8 + i]; }
  const float mu   = S * (1.0f / D_);
  const float var  = S2 * (1.0f / D_) - mu * mu;
  const float rstd = rsqrtf(var + 1e-5f);
  const float4 wv = reinterpret_cast<const float4*>(w)[t];
  const float4 bv = reinterpret_cast<const float4*>(bsh)[t];
  ushort4 ov;
  ov.x = f2bf((v.x - mu) * rstd * wv.x + bv.x);
  ov.y = f2bf((v.y - mu) * rstd * wv.y + bv.y);
  ov.z = f2bf((v.z - mu) * rstd * wv.z + bv.z);
  ov.w = f2bf((v.w - mu) * rstd * wv.w + bv.w);
  reinterpret_cast<ushort4*>(out + (size_t)row * D_)[t] = ov;
}

// ---------- generic WMMA GEMM : C[M,N] = epi(A_bf16[M,K] @ Bt_bf16[N,K]^T) ----------
#define BM 128
#define BN 128
#define BK 32
#define LDT 40  // padded LDS row stride (elems): 80B rows, 16B aligned, banks spread

enum { EPI_QKV = 0, EPI_OPROJ = 1, EPI_FF1 = 2, EPI_FF2 = 3 };

template <int EPI>
__global__ __launch_bounds__(256)
void gemm_kernel(const unsigned short* __restrict__ A,
                 const unsigned short* __restrict__ Bt,
                 const float* __restrict__ bias,
                 const float* __restrict__ resid,
                 float* __restrict__ outF,
                 unsigned short* __restrict__ outB,
                 unsigned short* __restrict__ vT,
                 int M, int N, int K) {
  __shared__ __align__(16) unsigned short As[2][BM * LDT];
  __shared__ __align__(16) unsigned short Bs[2][BN * LDT];
  const int tid  = threadIdx.x;
  const int lane = tid & 31;
  const int w    = tid >> 5;     // 8 waves
  const int wm   = w & 3;        // 4 row-granules of 32
  const int wn   = w >> 2;       // 2 col-granules of 64
  const int m0 = blockIdx.y * BM;
  const int n0 = blockIdx.x * BN;
  const int l16  = lane & 15;
  const int half = lane >> 4;
  const int ko   = half * 8;

  v8f acc[2][4];
#pragma unroll
  for (int i = 0; i < 2; ++i)
#pragma unroll
    for (int j = 0; j < 4; ++j) acc[i][j] = v8f_zero();

  const int ar0 = tid >> 2;   // 0..63 (row within pass)
  const int akg = tid & 3;    // k-group of 8 elems
  const int nk  = K / BK;

  // prefetch k-tile 0 into buffer 0
#pragma unroll
  for (int p = 0; p < 2; ++p) {
    const int r = p * 64 + ar0;
    cp16_g2l(&As[0][r * LDT + akg * 8],
             &A[(size_t)(m0 + r) * K + akg * 8]);
    cp16_g2l(&Bs[0][r * LDT + akg * 8],
             &Bt[(size_t)(n0 + r) * K + akg * 8]);
  }
  wait_async();
  __syncthreads();

  for (int kt = 0; kt < nk; ++kt) {
    const int cur = kt & 1;
    if (kt + 1 < nk) {  // prefetch next tile into the other buffer
      const int kb = (kt + 1) * BK;
#pragma unroll
      for (int p = 0; p < 2; ++p) {
        const int r = p * 64 + ar0;
        cp16_g2l(&As[cur ^ 1][r * LDT + akg * 8],
                 &A[(size_t)(m0 + r) * K + kb + akg * 8]);
        cp16_g2l(&Bs[cur ^ 1][r * LDT + akg * 8],
                 &Bt[(size_t)(n0 + r) * K + kb + akg * 8]);
      }
    }
    v16bf af[2], bf[4];
#pragma unroll
    for (int mt = 0; mt < 2; ++mt)
      af[mt] = load_frag(&As[cur][(wm * 32 + mt * 16 + l16) * LDT + ko]);
#pragma unroll
    for (int nt = 0; nt < 4; ++nt)
      bf[nt] = load_frag(&Bs[cur][(wn * 64 + nt * 16 + l16) * LDT + ko]);
#pragma unroll
    for (int mt = 0; mt < 2; ++mt)
#pragma unroll
      for (int nt = 0; nt < 4; ++nt)
        acc[mt][nt] = wmma_bf16(af[mt], bf[nt], acc[mt][nt]);
    wait_async();
    __syncthreads();
  }

  // epilogue: C element (i,lane) -> row = i + 8*(lane/16), col = lane%16
#pragma unroll
  for (int mt = 0; mt < 2; ++mt) {
#pragma unroll
    for (int nt = 0; nt < 4; ++nt) {
      const int n = n0 + wn * 64 + nt * 16 + l16;
      const float bv = bias[n];
#pragma unroll
      for (int i = 0; i < 8; ++i) {
        const int m = m0 + wm * 32 + mt * 16 + i + half * 8;
        const float v = acc[mt][nt][i] + bv;
        if constexpr (EPI == EPI_QKV) {
          if (n < 2 * D_) {
            outB[(size_t)m * (3 * D_) + n] = f2bf(v);  // q,k row-major
          } else {                                     // v stored transposed [b][h][d][t]
            const int hd = n - 2 * D_;
            const int hh = hd >> 6, dd = hd & 63;
            const int bb = m >> 11, tt = m & (T_ - 1);
            vT[(((size_t)(bb * H_ + hh)) * HD_ + dd) * T_ + tt] = f2bf(v);
          }
        } else if constexpr (EPI == EPI_OPROJ) {
          outF[(size_t)m * N + n] = resid[(size_t)m * N + n] + v;
        } else if constexpr (EPI == EPI_FF1) {
          const float g =
              0.5f * v * (1.0f + tanhf(0.7978845608028654f * (v + 0.044715f * v * v * v)));
          outB[(size_t)m * N + n] = f2bf(g);
        } else {  // EPI_FF2
          outF[(size_t)m * N + n] = resid[(size_t)m * N + n] + v;
        }
      }
    }
  }
}

// ---------- causal flash attention (bf16 WMMA, f32 online softmax) ----------
#define ATT_BQ 128
#define ATT_BK 128
#define QLD 72    // 144B rows: 16B aligned, banks spread
#define PLD 136   // 272B rows: 16B aligned, conflict-free frag reads

__global__ __launch_bounds__(256)
void attn_kernel(const unsigned short* __restrict__ qkv,
                 const unsigned short* __restrict__ vT,
                 unsigned short* __restrict__ av) {
  __shared__ __align__(16) unsigned short Qs[ATT_BQ * QLD];
  __shared__ __align__(16) unsigned short Ks[2][ATT_BK * QLD];
  __shared__ __align__(16) unsigned short Vt[2][HD_ * PLD];
  __shared__ __align__(16) unsigned short Ps[ATT_BQ * PLD];

  const int tid  = threadIdx.x;
  const int lane = tid & 31;
  const int w    = tid >> 5;       // 8 waves, each owns 16 q-rows
  const int qt = blockIdx.x;       // q tile 0..15
  const int h  = blockIdx.y;
  const int b  = blockIdx.z;
  const int q_base = qt * ATT_BQ;
  const int l16  = lane & 15;
  const int half = lane >> 4;
  const int ko   = half * 8;
  const int wr0  = w * 16;

  const size_t qkv_row = 3 * D_;
  const unsigned short* qbase = qkv + ((size_t)b * T_) * qkv_row + h * HD_;
  const unsigned short* kbase = qbase + D_;
  const unsigned short* vtb   = vT + ((size_t)(b * H_ + h)) * HD_ * T_;

  // K tile (128 x 64, [kk][d]) + V tile (transposed in global: [d][kk])
  auto issue_kv = [&](int j, int buf) {
#pragma unroll
    for (int p = 0; p < 4; ++p) {
      const int idx = p * 256 + tid;
      const int r = idx >> 3, ck = idx & 7;
      cp16_g2l(&Ks[buf][r * QLD + ck * 8],
               &kbase[(size_t)(j * ATT_BK + r) * qkv_row + ck * 8]);
    }
#pragma unroll
    for (int p = 0; p < 4; ++p) {
      const int idx = p * 256 + tid;
      const int d = idx >> 4, ck = idx & 15;
      cp16_g2l(&Vt[buf][d * PLD + ck * 8],
               &vtb[(size_t)d * T_ + j * ATT_BK + ck * 8]);
    }
  };

  // load Q tile (128 x 64) and first K/V tile
#pragma unroll
  for (int p = 0; p < 4; ++p) {
    const int idx = p * 256 + tid;
    const int r = idx >> 3, ck = idx & 7;
    cp16_g2l(&Qs[r * QLD + ck * 8],
             &qbase[(size_t)(q_base + r) * qkv_row + ck * 8]);
  }
  issue_kv(0, 0);
  wait_async();
  __syncthreads();

  float mrow[8], lrow[8];
  v8f o[4];
#pragma unroll
  for (int i = 0; i < 8; ++i) { mrow[i] = -3.0e38f; lrow[i] = 0.0f; }
#pragma unroll
  for (int dt = 0; dt < 4; ++dt) o[dt] = v8f_zero();

  for (int j = 0; j <= qt; ++j) {
    const int cur = j & 1;
    if (j < qt) issue_kv(j + 1, cur ^ 1);  // prefetch while computing

    // S = Q K^T : this wave's 16 rows x 128 cols
    v8f s[8];
#pragma unroll
    for (int nt = 0; nt < 8; ++nt) s[nt] = v8f_zero();
#pragma unroll
    for (int c = 0; c < 2; ++c) {
      const v16bf aq = load_frag(&Qs[(wr0 + l16) * QLD + c * 32 + ko]);
#pragma unroll
      for (int nt = 0; nt < 8; ++nt) {
        const v16bf bk = load_frag(&Ks[cur][(nt * 16 + l16) * QLD + c * 32 + ko]);
        s[nt] = wmma_bf16(aq, bk, s[nt]);
      }
    }
    // scale + causal mask on diagonal tile
    const bool diag = (j == qt);
#pragma unroll
    for (int nt = 0; nt < 8; ++nt) {
#pragma unroll
      for (int i = 0; i < 8; ++i) {
        float sv = s[nt][i] * 0.125f;  // 1/sqrt(64)
        if (diag) {
          const int col = j * ATT_BK + nt * 16 + l16;
          const int row = q_base + wr0 + i + half * 8;
          if (col > row) sv = -1.0e30f;
        }
        s[nt][i] = sv;
      }
    }
    // row max (row lives across a 16-lane half at fixed accum index i)
    float rsum[8];
#pragma unroll
    for (int i = 0; i < 8; ++i) {
      float v = s[0][i];
#pragma unroll
      for (int nt = 1; nt < 8; ++nt) v = fmaxf(v, s[nt][i]);
      v = fmaxf(v, __shfl_xor(v, 1, 32));
      v = fmaxf(v, __shfl_xor(v, 2, 32));
      v = fmaxf(v, __shfl_xor(v, 4, 32));
      v = fmaxf(v, __shfl_xor(v, 8, 32));
      const float mnew = fmaxf(mrow[i], v);
      const float sc   = __expf(mrow[i] - mnew);
      mrow[i] = mnew;
      lrow[i] *= sc;
#pragma unroll
      for (int dt = 0; dt < 4; ++dt) o[dt][i] *= sc;
      rsum[i] = 0.f;
    }
    // P = exp(S - m) -> LDS (each wave writes only its own 16 rows)
#pragma unroll
    for (int nt = 0; nt < 8; ++nt) {
#pragma unroll
      for (int i = 0; i < 8; ++i) {
        const float p = __expf(s[nt][i] - mrow[i]);
        rsum[i] += p;
        Ps[(wr0 + i + half * 8) * PLD + nt * 16 + l16] = f2bf(p);
      }
    }
#pragma unroll
    for (int i = 0; i < 8; ++i) {
      float v = rsum[i];
      v += __shfl_xor(v, 1, 32);
      v += __shfl_xor(v, 2, 32);
      v += __shfl_xor(v, 4, 32);
      v += __shfl_xor(v, 8, 32);
      lrow[i] += v;
    }
    __syncthreads();  // order P stores before fragment reads
    // O += P V
#pragma unroll
    for (int c4 = 0; c4 < 4; ++c4) {
      const v16bf ap = load_frag(&Ps[(wr0 + l16) * PLD + c4 * 32 + ko]);
#pragma unroll
      for (int dt = 0; dt < 4; ++dt) {
        const v16bf bv = load_frag(&Vt[cur][(dt * 16 + l16) * PLD + c4 * 32 + ko]);
        o[dt] = wmma_bf16(ap, bv, o[dt]);
      }
    }
    wait_async();     // prefetched K/V tile has landed
    __syncthreads();  // all waves done reading this tile's buffers
  }

  // normalize + store av (bf16, row-major [token][D])
#pragma unroll
  for (int i = 0; i < 8; ++i) {
    const float inv = 1.0f / lrow[i];
    const int row = q_base + wr0 + i + half * 8;
    const size_t obase = ((size_t)(b * T_) + row) * D_ + h * HD_;
#pragma unroll
    for (int dt = 0; dt < 4; ++dt)
      av[obase + dt * 16 + l16] = f2bf(o[dt][i] * inv);
  }
}

// ---------- host launcher ----------
extern "C" void kernel_launch(void* const* d_in, const int* in_sizes, int n_in,
                              void* d_out, int out_size, void* d_ws, size_t ws_size,
                              hipStream_t stream) {
  (void)in_sizes; (void)n_in; (void)out_size; (void)ws_size;
  const float* x     = (const float*)d_in[0];
  const float* ln1_w = (const float*)d_in[1];
  const float* ln1_b = (const float*)d_in[2];
  const float* W_qkv = (const float*)d_in[3];
  const float* b_qkv = (const float*)d_in[4];
  const float* W_o   = (const float*)d_in[5];
  const float* b_o   = (const float*)d_in[6];
  const float* ln2_w = (const float*)d_in[7];
  const float* ln2_b = (const float*)d_in[8];
  const float* W_ff1 = (const float*)d_in[9];
  const float* b_ff1 = (const float*)d_in[10];
  const float* W_ff2 = (const float*)d_in[11];
  const float* b_ff2 = (const float*)d_in[12];
  float* out = (float*)d_out;
  char* ws = (char*)d_ws;

  // workspace layout (bytes)
  constexpr size_t off_wqkv = 0;                               // 3072x1024 bf16
  constexpr size_t off_wo   = off_wqkv + (size_t)3072 * 1024 * 2;
  constexpr size_t off_wff1 = off_wo   + (size_t)1024 * 1024 * 2;
  constexpr size_t off_wff2 = off_wff1 + (size_t)4096 * 1024 * 2;
  constexpr size_t off_h    = off_wff2 + (size_t)1024 * 4096 * 2;  // h / h2 (aliased)
  constexpr size_t off_qkv  = off_h    + (size_t)NTOK * D_ * 2;    // q,k rows (v slot unused)
  constexpr size_t off_vT   = off_qkv  + (size_t)NTOK * 3 * D_ * 2;
  constexpr size_t off_ff1  = off_qkv;                             // reuses qkv+vT after attn
  constexpr size_t off_av   = off_vT   + (size_t)NTOK * D_ * 2;
  constexpr size_t off_x1   = off_av   + (size_t)NTOK * D_ * 2;

  unsigned short* wqkv_t = (unsigned short*)(ws + off_wqkv);
  unsigned short* wo_t   = (unsigned short*)(ws + off_wo);
  unsigned short* wff1_t = (unsigned short*)(ws + off_wff1);
  unsigned short* wff2_t = (unsigned short*)(ws + off_wff2);
  unsigned short* hbuf   = (unsigned short*)(ws + off_h);
  unsigned short* qkvb   = (unsigned short*)(ws + off_qkv);
  unsigned short* vTb    = (unsigned short*)(ws + off_vT);
  unsigned short* ff1b   = (unsigned short*)(ws + off_ff1);
  unsigned short* avb    = (unsigned short*)(ws + off_av);
  float*          x1     = (float*)(ws + off_x1);

  const dim3 t32x8(32, 8);
  transpose_f32_bf16<<<dim3(3072 / 32, 1024 / 32), t32x8, 0, stream>>>(W_qkv, wqkv_t, 1024, 3072);
  transpose_f32_bf16<<<dim3(1024 / 32, 1024 / 32), t32x8, 0, stream>>>(W_o,   wo_t,   1024, 1024);
  transpose_f32_bf16<<<dim3(4096 / 32, 1024 / 32), t32x8, 0, stream>>>(W_ff1, wff1_t, 1024, 4096);
  transpose_f32_bf16<<<dim3(1024 / 32, 4096 / 32), t32x8, 0, stream>>>(W_ff2, wff2_t, 4096, 1024);

  ln_kernel<<<NTOK, 256, 0, stream>>>(x, ln1_w, ln1_b, hbuf);

  gemm_kernel<EPI_QKV><<<dim3(3072 / BN, NTOK / BM), 256, 0, stream>>>(
      hbuf, wqkv_t, b_qkv, nullptr, nullptr, qkvb, vTb, NTOK, 3072, 1024);

  attn_kernel<<<dim3(T_ / ATT_BQ, H_, B_), 256, 0, stream>>>(qkvb, vTb, avb);

  gemm_kernel<EPI_OPROJ><<<dim3(1024 / BN, NTOK / BM), 256, 0, stream>>>(
      avb, wo_t, b_o, x, x1, nullptr, nullptr, NTOK, 1024, 1024);

  ln_kernel<<<NTOK, 256, 0, stream>>>(x1, ln2_w, ln2_b, hbuf);

  gemm_kernel<EPI_FF1><<<dim3(4096 / BN, NTOK / BM), 256, 0, stream>>>(
      hbuf, wff1_t, b_ff1, nullptr, nullptr, ff1b, nullptr, NTOK, 4096, 1024);

  gemm_kernel<EPI_FF2><<<dim3(1024 / BN, NTOK / BM), 256, 0, stream>>>(
      ff1b, wff2_t, b_ff2, x1, out, nullptr, nullptr, NTOK, 1024, 4096);
}
